// SeriesAttn1_18760417149546
// MI455X (gfx1250) — matrix operand order repaired and checked by reading, not compiled, not run on hardware
//
#include <hip/hip_runtime.h>
#include <hip/hip_bf16.h>

#define LSEQ 4096
#define EDIM 512
#define QD   96
#define HD   32
#define NH   3
#define TOPK 20
#define CHUNK 128

typedef __attribute__((ext_vector_type(16))) __bf16 v16bf;
typedef __attribute__((ext_vector_type(8)))  float  v8f;
typedef __attribute__((ext_vector_type(4)))  unsigned int u32x4;
typedef __attribute__((ext_vector_type(8)))  int   i32x8;
typedef __attribute__((ext_vector_type(4)))  int   i32x4;

#if defined(__has_builtin)
#if __has_builtin(__builtin_amdgcn_tensor_load_to_lds)
#define HAVE_TDM 1
#endif
#endif

union Frag { v16bf v; uint4 u[2]; };

// A-matrix 16x32 bf16 fragment: lane row = lane&15,
// contiguous K range [kb, kb+16), kb = (lane>>4)*16.
__device__ __forceinline__ v16bf load_fragA(const __bf16* rowptr, int lane, int k0) {
    int kb = k0 + ((lane >> 4) << 4);
    Frag f;
    f.u[0] = *(const uint4*)(rowptr + kb);
    f.u[1] = *(const uint4*)(rowptr + kb + 8);
    return f.v;
}

// B-matrix 32x16 bf16 fragment: lane col = lane&15, contiguous K range
// [hi*16, hi*16+16) per lane (col n of B == row n of source, K-contiguous).
__device__ __forceinline__ v16bf load_fragB(const __bf16* rowptr, int lane, int k0) {
    int kb = k0 + ((lane >> 4) << 4);
    Frag f;
    f.u[0] = *(const uint4*)(rowptr + kb);
    f.u[1] = *(const uint4*)(rowptr + kb + 8);
    return f.v;
}

#ifdef HAVE_TDM
// TDM: 1-D contiguous copy of CHUNK*HD bf16 (8 KB) from global to LDS.
// D# per ISA ch8: group0 {count=1 | lds_addr | global_addr | type=2},
// group1 {data_size=2B, tensor_dim0=1<<30 (no OOB), tile_dim0=4096 elem,
//         tile_dim1/2 unused, dim0_stride=4096}.
__device__ __forceinline__ void tdm_load_chunk(const __bf16* gsrc, unsigned int lds_off) {
    unsigned long long ga = (unsigned long long)(uintptr_t)gsrc;
    u32x4 g0;
    g0[0] = 1u;                                           // count=1, user descriptor
    g0[1] = lds_off;                                      // LDS byte address
    g0[2] = (unsigned int)ga;                             // global_addr[31:0]
    g0[3] = (unsigned int)((ga >> 32) & 0x1FFFFFFu) | (2u << 30); // [56:32] | type=2
    i32x8 g1;
    g1[0] = 1 << 16;                                      // data_size = 2 bytes
    g1[1] = 0;                                            // tensor_dim0[15:0] (dim0=1<<30)
    g1[2] = 0x4000;                                       // tensor_dim0[31:16]
    g1[3] = (int)((unsigned int)(CHUNK * HD) << 16);      // tile_dim0 = 4096 elements
    g1[4] = 0;                                            // tile_dim1/2 unused (1-D)
    g1[5] = CHUNK * HD;                                   // tensor_dim0_stride
    g1[6] = 0;
    g1[7] = 0;
    i32x4 z4 = {0, 0, 0, 0};
#if __clang_major__ >= 23
    i32x8 z8 = {0, 0, 0, 0, 0, 0, 0, 0};
    __builtin_amdgcn_tensor_load_to_lds(g0, g1, z4, z4, z8, 0);
#else
    __builtin_amdgcn_tensor_load_to_lds(g0, g1, z4, z4, 0);
#endif
}
#endif

// ---------------------------------------------------------------- K1: fp32 -> bf16
__global__ void k_convert(const float* __restrict__ x, const float* __restrict__ win,
                          const float* __restrict__ wo,
                          __bf16* __restrict__ xb, __bf16* __restrict__ winb,
                          __bf16* __restrict__ wob) {
    const int nx = LSEQ * EDIM;
    const int nw = 3 * QD * EDIM;
    const int no = EDIM * QD;
    int stride = gridDim.x * blockDim.x;
    for (int i = blockIdx.x * blockDim.x + threadIdx.x; i < nx + nw + no; i += stride) {
        if (i < nx)            xb[i]            = (__bf16)x[i];
        else if (i < nx + nw)  winb[i - nx]     = (__bf16)win[i - nx];
        else                   wob[i - nx - nw] = (__bf16)wo[i - nx - nw];
    }
}

// ---------------------------------------------------------------- K2: QKV projection
__global__ __launch_bounds__(256) void k_qkv(const __bf16* __restrict__ xb,
                                             const __bf16* __restrict__ winb,
                                             const float* __restrict__ bias,
                                             __bf16* __restrict__ qp,
                                             __bf16* __restrict__ kp,
                                             float* __restrict__ vp) {
    const int NT = (3 * QD) / 16;                              // 18 col tiles
    int wave = (blockIdx.x * blockDim.x + threadIdx.x) >> 5;
    int lane = threadIdx.x & 31;
    if (wave >= (LSEQ / 16) * NT) return;                      // wave-uniform guard
    int m0 = (wave / NT) * 16;
    int n0 = (wave % NT) * 16;
    const __bf16* arow = xb   + (size_t)(m0 + (lane & 15)) * EDIM;
    const __bf16* brow = winb + (size_t)(n0 + (lane & 15)) * EDIM;
    v8f acc = {};
    for (int k0 = 0; k0 < EDIM; k0 += 32) {
        v16bf a = load_fragA(arow, lane, k0);
        v16bf b = load_fragB(brow, lane, k0);
        acc = __builtin_amdgcn_wmma_f32_16x16x32_bf16(false, a, false, b,
                                                      (short)0, acc, false, false);
    }
    int col  = n0 + (lane & 15);
    int hi   = lane >> 4;
    float bv = bias[col];
    int sec  = col / QD;            // 0=q 1=k 2=v
    int w96  = col % QD;
    int head = w96 >> 5, dim = w96 & 31;
    const float rs = 0.17677669529663687f;   // 1/sqrt(32)
    #pragma unroll
    for (int i = 0; i < 8; ++i) {
        int row = m0 + i + (hi << 3);
        float v = acc[i] + bv;
        size_t off = ((size_t)head * LSEQ + row) * HD + dim;
        if (sec == 0)      qp[off] = (__bf16)(v * rs);
        else if (sec == 1) kp[off] = (__bf16)v;
        else               vp[off] = v;
    }
}

// ---------------------------------------------------------------- K3: fused attention
// per block: 16 rows of one head. TDM double-buffers K chunks into LDS,
// WMMA scores from LDS fragments, private top-20, merge, softmax, sparse A*V.
__global__ __launch_bounds__(256) void k_attn(const __bf16* __restrict__ qp,
                                              const __bf16* __restrict__ kp,
                                              const float* __restrict__ vp,
                                              __bf16* __restrict__ ao) {
    // LDS overlay (42.5 KB total):
    //  phase 1 (chunk loop): [0,16K) K stage x2, [16K,24K) score chunk
    //  phase 2 (merge):      [0,20K) cand vals,  [20K,40K) cand idx
    //  tail (always):        [40K, ..) final top-20 vals/idx
    __shared__ __align__(16) char smem[43520];
    __bf16* kstage = (__bf16*)smem;                            // [2][CHUNK][HD]
    float (*s_sc)[CHUNK] = (float (*)[CHUNK])(smem + 16384);   // [16][CHUNK]
    float* s_cv = (float*)smem;                                // [16][16*TOPK]
    int*   s_ci = (int*)(smem + 20480);                        // [16][16*TOPK]
    float (*s_fv)[TOPK] = (float (*)[TOPK])(smem + 40960);
    int   (*s_fi)[TOPK] = (int   (*)[TOPK])(smem + 42240);

    int h  = blockIdx.x / (LSEQ / 16);
    int r0 = (blockIdx.x % (LSEQ / 16)) * 16;
    int tid  = threadIdx.x;
    int wv   = tid >> 5, lane = tid & 31;
    int hi   = lane >> 4, lrow = lane & 15;
    int myrow = tid >> 4, mslot = tid & 15;

    const __bf16* qbase = qp + (size_t)h * LSEQ * HD;
    const __bf16* kbase = kp + (size_t)h * LSEQ * HD;
    const float*  vbase = vp + (size_t)h * LSEQ * HD;

    // loop-invariant A fragment: q rows r0..r0+15, full K=32 (head_dim)
    v16bf afr = load_fragA(qbase + (size_t)(r0 + lrow) * HD, lane, 0);

    float pv[TOPK]; int pi[TOPK];
    #pragma unroll
    for (int i = 0; i < TOPK; ++i) { pv[i] = -INFINITY; pi[i] = -1; }
    float pmin = -INFINITY; int pslot = 0;

#ifdef HAVE_TDM
    unsigned int stage_off = (unsigned int)(uintptr_t)kstage;
    if (wv == 0) tdm_load_chunk(kbase, stage_off);             // prologue: chunk 0 -> buf 0
#endif

    int buf = 0;
    for (int c0 = 0; c0 < LSEQ; c0 += CHUNK, buf ^= 1) {
#ifdef HAVE_TDM
        if (wv == 0) {
            if (c0 + CHUNK < LSEQ) {
                tdm_load_chunk(kbase + (size_t)(c0 + CHUNK) * HD,
                               stage_off + (unsigned int)((buf ^ 1) * CHUNK * HD * 2));
                __builtin_amdgcn_s_wait_tensorcnt(1);          // buf ready, next in flight
            } else {
                __builtin_amdgcn_s_wait_tensorcnt(0);
            }
        }
#else
        {   // cooperative fallback copy: 8 KB = 512 uint4, 2 per thread
            const uint4* src = (const uint4*)(kbase + (size_t)c0 * HD);
            uint4* dst = (uint4*)(kstage + (size_t)buf * CHUNK * HD);
            dst[tid]       = src[tid];
            dst[tid + 256] = src[tid + 256];
        }
#endif
        __syncthreads();                                       // stage[buf] visible

        // score tile: wave wv -> chunk cols [wv*16, wv*16+16), B frags from LDS
        {
            const __bf16* srow = kstage + (size_t)buf * CHUNK * HD
                               + (size_t)((wv << 4) + lrow) * HD + (hi << 4);
            Frag f;
            f.u[0] = *(const uint4*)srow;                      // ds_load_b128
            f.u[1] = *(const uint4*)(srow + 8);
            v8f acc = {};
            acc = __builtin_amdgcn_wmma_f32_16x16x32_bf16(false, afr, false, f.v,
                                                          (short)0, acc, false, false);
            #pragma unroll
            for (int i = 0; i < 8; ++i)
                s_sc[i + (hi << 3)][(wv << 4) + lrow] = acc[i];
        }
        __syncthreads();                                       // scores ready

        // selection: thread (myrow, mslot) scans 8 of the 128 chunk columns
        for (int j = mslot; j < CHUNK; j += 16) {
            float v = s_sc[myrow][j];
            if (v > pmin) {
                pv[pslot] = v; pi[pslot] = c0 + j;
                float mn = pv[0]; int ms = 0;
                #pragma unroll
                for (int q = 1; q < TOPK; ++q)
                    if (pv[q] < mn) { mn = pv[q]; ms = q; }
                pmin = mn; pslot = ms;
            }
        }
        __syncthreads();    // selection done: s_sc free + stage[buf] retire-able
    }

    // dump private lists for final merge (overlays stage/score LDS)
    #pragma unroll
    for (int i = 0; i < TOPK; ++i) {
        s_cv[myrow * (16 * TOPK) + mslot * TOPK + i] = pv[i];
        s_ci[myrow * (16 * TOPK) + mslot * TOPK + i] = pi[i];
    }
    __syncthreads();

    // row owners: top-20 of 320 candidates, then softmax over the 20
    if (tid < 16) {
        int row = tid;
        float fv[TOPK]; int fi[TOPK];
        #pragma unroll
        for (int i = 0; i < TOPK; ++i) { fv[i] = -INFINITY; fi[i] = -1; }
        float mn = -INFINITY; int ms = 0;
        for (int j = 0; j < 16 * TOPK; ++j) {
            float v = s_cv[row * (16 * TOPK) + j];
            if (v > mn) {
                fv[ms] = v; fi[ms] = s_ci[row * (16 * TOPK) + j];
                mn = fv[0]; ms = 0;
                for (int q = 1; q < TOPK; ++q)
                    if (fv[q] < mn) { mn = fv[q]; ms = q; }
            }
        }
        float m = -INFINITY;
        for (int i = 0; i < TOPK; ++i) m = fmaxf(m, fv[i]);
        float s = 0.f;
        for (int i = 0; i < TOPK; ++i) { fv[i] = __expf(fv[i] - m); s += fv[i]; }
        float inv = 1.f / s;
        for (int i = 0; i < TOPK; ++i) { s_fv[row][i] = fv[i] * inv; s_fi[row][i] = fi[i]; }
    }
    __syncthreads();

    // sparse A*V: thread -> (row, 2 output dims)
    {
        int d0 = mslot * 2;
        float o0 = 0.f, o1 = 0.f;
        #pragma unroll
        for (int i = 0; i < TOPK; ++i) {
            float p  = s_fv[myrow][i];
            int  idx = s_fi[myrow][i];
            const float* vr = vbase + (size_t)idx * HD;
            o0 += p * vr[d0];
            o1 += p * vr[d0 + 1];
        }
        size_t o = (size_t)(r0 + myrow) * QD + h * HD + d0;
        ao[o]     = (__bf16)o0;
        ao[o + 1] = (__bf16)o1;
    }
}

// ---------------------------------------------------------------- K4: output proj
__global__ __launch_bounds__(256) void k_outproj(const __bf16* __restrict__ ao,
                                                 const __bf16* __restrict__ wob,
                                                 const float* __restrict__ ob,
                                                 float* __restrict__ out) {
    const int NT = EDIM / 16;                                  // 32 col tiles
    int wave = (blockIdx.x * blockDim.x + threadIdx.x) >> 5;
    int lane = threadIdx.x & 31;
    if (wave >= (LSEQ / 16) * NT) return;
    int m0 = (wave / NT) * 16;
    int n0 = (wave % NT) * 16;
    const __bf16* arow = ao  + (size_t)(m0 + (lane & 15)) * QD;
    const __bf16* brow = wob + (size_t)(n0 + (lane & 15)) * QD;
    __builtin_prefetch(arow, 0, 0);                            // global_prefetch_b8
    v8f acc = {};
    #pragma unroll
    for (int k0 = 0; k0 < QD; k0 += 32) {
        v16bf a = load_fragA(arow, lane, k0);
        v16bf b = load_fragB(brow, lane, k0);
        acc = __builtin_amdgcn_wmma_f32_16x16x32_bf16(false, a, false, b,
                                                      (short)0, acc, false, false);
    }
    int col = n0 + (lane & 15);
    int hi  = lane >> 4;
    float bv = ob[col];
    #pragma unroll
    for (int i = 0; i < 8; ++i) {
        int row = m0 + i + (hi << 3);
        out[(size_t)row * EDIM + col] = acc[i] + bv;
    }
}

// ---------------------------------------------------------------- launch
extern "C" void kernel_launch(void* const* d_in, const int* in_sizes, int n_in,
                              void* d_out, int out_size, void* d_ws, size_t ws_size,
                              hipStream_t stream) {
    const float* x    = (const float*)d_in[0];
    const float* win  = (const float*)d_in[1];
    const float* bias = (const float*)d_in[2];
    const float* outw = (const float*)d_in[3];
    const float* outb = (const float*)d_in[4];
    float* out = (float*)d_out;

    char* ws = (char*)d_ws;
    size_t off = 0;
    auto alloc = [&](size_t bytes) -> char* {
        char* p = ws + off;
        off = (off + bytes + 255) & ~(size_t)255;
        return p;
    };
    __bf16* xb   = (__bf16*)alloc((size_t)LSEQ * EDIM * 2);
    __bf16* winb = (__bf16*)alloc((size_t)3 * QD * EDIM * 2);
    __bf16* wob  = (__bf16*)alloc((size_t)EDIM * QD * 2);
    __bf16* qp   = (__bf16*)alloc((size_t)NH * LSEQ * HD * 2);
    __bf16* kp   = (__bf16*)alloc((size_t)NH * LSEQ * HD * 2);
    float*  vp   = (float*) alloc((size_t)NH * LSEQ * HD * 4);
    __bf16* ao   = (__bf16*)alloc((size_t)LSEQ * QD * 2);

    k_convert<<<512, 256, 0, stream>>>(x, win, outw, xb, winb, wob);
    k_qkv<<<576, 256, 0, stream>>>(xb, winb, bias, qp, kp, vp);
    k_attn<<<NH * (LSEQ / 16), 256, 0, stream>>>(qp, kp, vp, ao);
    k_outproj<<<1024, 256, 0, stream>>>(ao, wob, outb, out);
}
